// ContraCLMTokLoss_51084341019237
// MI455X (gfx1250) — matrix operand order repaired and checked by reading, not compiled, not run on hardware
//
#include <hip/hip_runtime.h>

// ContraCLM token-level contrastive loss, fused for MI455X (gfx1250).
// Gram matrix X·X^T per batch (3072x1024 bf16) via v_wmma_f32_16x16x32_bf16,
// fused exp/mask/row-sum epilogue (sim tensor never materialized), with
// double-buffered GLOBAL_LOAD_ASYNC_TO_LDS_B128 staging (ASYNCcnt pipeline).

#define B_    8
#define S_    1536
#define N2_   3072        // 2*S
#define D_    1024
#define INV_T 20.0f       // 1/temperature
#define KC_   64          // K chunk per pipeline stage
#define ASTR  72          // LDS row stride in bf16 elems (64 + 8 pad): 144B rows
#define NPANEL (N2_ / 64) // 48 column panels
#define NKC    (D_ / KC_) // 16 k-chunks per panel

typedef __bf16 v16bf __attribute__((ext_vector_type(16)));
typedef float  v8f   __attribute__((ext_vector_type(8)));

union Frag { uint4 q[2]; v16bf v; };

__device__ __forceinline__ unsigned short f2bf(float f) {
  unsigned u = __float_as_uint(f);
  u += 0x7FFFu + ((u >> 16) & 1u);   // round-to-nearest-even
  return (unsigned short)(u >> 16);
}

__device__ __forceinline__ void async_b128(const unsigned short* lds,
                                           const unsigned short* glob) {
  unsigned           l = (unsigned)(uintptr_t)lds;        // addr[31:0] = LDS offset
  unsigned long long g = (unsigned long long)(uintptr_t)glob;
  asm volatile("global_load_async_to_lds_b128 %0, %1, off"
               :: "v"(l), "v"(g) : "memory");
}

// ---------------------------------------------------------------------------
// Kernel 1: L2-normalize both views -> bf16 X[b][0..2S)[D], store pos_sim/T.
// ---------------------------------------------------------------------------
__global__ __launch_bounds__(256) void norm_kernel(
    const float* __restrict__ h1, const float* __restrict__ h2,
    unsigned short* __restrict__ Xb, float* __restrict__ possim) {
  const int s = blockIdx.x;
  const int b = blockIdx.y;
  const int t = threadIdx.x;
  const size_t base = ((size_t)b * S_ + s) * D_;

  float4 a = *(const float4*)(h1 + base + t * 4);
  float4 c = *(const float4*)(h2 + base + t * 4);
  float s1 = a.x * a.x + a.y * a.y + a.z * a.z + a.w * a.w;
  float s2 = c.x * c.x + c.y * c.y + c.z * c.z + c.w * c.w;
  float dd = a.x * c.x + a.y * c.y + a.z * c.z + a.w * c.w;

  #pragma unroll
  for (int off = 16; off; off >>= 1) {
    s1 += __shfl_xor(s1, off);
    s2 += __shfl_xor(s2, off);
    dd += __shfl_xor(dd, off);
  }
  __shared__ float red[3][8];
  __shared__ float bc[2];
  const int w = t >> 5, l = t & 31;
  if (l == 0) { red[0][w] = s1; red[1][w] = s2; red[2][w] = dd; }
  __syncthreads();
  if (t == 0) {
    float t1 = 0.f, t2 = 0.f, t3 = 0.f;
    #pragma unroll
    for (int i = 0; i < 8; ++i) { t1 += red[0][i]; t2 += red[1][i]; t3 += red[2][i]; }
    float i1 = 1.0f / fmaxf(sqrtf(t1), 1e-12f);
    float i2 = 1.0f / fmaxf(sqrtf(t2), 1e-12f);
    bc[0] = i1; bc[1] = i2;
    possim[(size_t)b * S_ + s] = t3 * i1 * i2 * INV_T;
  }
  __syncthreads();
  const float i1 = bc[0], i2 = bc[1];

  ushort4 o1, o2;
  o1.x = f2bf(a.x * i1); o1.y = f2bf(a.y * i1); o1.z = f2bf(a.z * i1); o1.w = f2bf(a.w * i1);
  o2.x = f2bf(c.x * i2); o2.y = f2bf(c.y * i2); o2.z = f2bf(c.z * i2); o2.w = f2bf(c.w * i2);
  const size_t ob = ((size_t)b * N2_ + s) * D_ + (size_t)t * 4;
  *(ushort4*)(Xb + ob) = o1;                         // view-1 row s
  *(ushort4*)(Xb + ob + (size_t)S_ * D_) = o2;       // view-2 row S+s
}

// ---------------------------------------------------------------------------
// Kernel 2: fused Gram-matrix + exp + mask + row-sum, double-buffered async
// staging. Grid (N2/64, B), block = 128 threads (4 waves), wave owns 16 rows.
// Ng[b][i] = sum_j mask[j%S] * (j%S != i%S) * exp(x_i . x_j / T)
// ---------------------------------------------------------------------------
__global__ __launch_bounds__(128) void gemm_rowsum_kernel(
    const unsigned short* __restrict__ Xb,
    const unsigned char* __restrict__ mask,
    float* __restrict__ Ng) {
  __shared__ __align__(16) unsigned short sA[2][64 * ASTR];
  __shared__ __align__(16) unsigned short sB[2][64 * ASTR];

  const int b      = blockIdx.y;
  const int rowBlk = blockIdx.x * 64;
  const int tid    = threadIdx.x;
  const int wave   = tid >> 5;
  const int lane   = tid & 31;
  const int m      = lane & 15;   // row (A) / col (B,C) within 16x16 tile
  const int hl     = lane >> 4;   // lane half

  const unsigned short* X  = Xb + (size_t)b * N2_ * D_;
  const unsigned char*  mb = mask + (size_t)b * S_;
  const int rowStart = rowBlk + wave * 16;

  float racc[8] = {0.f, 0.f, 0.f, 0.f, 0.f, 0.f, 0.f, 0.f};

  // Stage one (jo, kc) chunk: A rows [rowBlk,+64), B cols [jo*64,+64), KC_ k.
  auto stage = [&](int buf, int jo, int kc) {
    #pragma unroll
    for (int u = 0; u < 4; ++u) {
      const int tt  = tid + u * 128;
      const int r   = tt >> 3;        // 0..63
      const int seg = tt & 7;         // 8 bf16 per b128
      async_b128(&sA[buf][r * ASTR + seg * 8],
                 X + (size_t)(rowBlk + r) * D_ + kc + seg * 8);
      async_b128(&sB[buf][r * ASTR + seg * 8],
                 X + (size_t)(jo * 64 + r) * D_ + kc + seg * 8);
    }
  };

  stage(0, 0, 0);   // prime the pipeline

  for (int jo = 0; jo < NPANEL; ++jo) {
    v8f acc[4];
    #pragma unroll
    for (int jj = 0; jj < 4; ++jj)
      #pragma unroll
      for (int r = 0; r < 8; ++r) acc[jj][r] = 0.f;

    // NKC is even, so buffer parity within a panel is just kcIdx&1 and every
    // panel starts on buffer 0.
    #pragma unroll 2
    for (int kcIdx = 0; kcIdx < NKC; ++kcIdx) {
      const int buf = kcIdx & 1;

      // Wait for this chunk's DMA; barrier makes it visible to all waves and
      // guarantees everyone is done with the buffer we prefetch into next.
      asm volatile("s_wait_asynccnt 0x0" ::: "memory");
      __syncthreads();

      if (kcIdx + 1 < NKC)            stage(buf ^ 1, jo, (kcIdx + 1) * KC_);
      else if (jo + 1 < NPANEL)       stage(buf ^ 1, jo + 1, 0);

      const unsigned short* cA = sA[buf];
      const unsigned short* cB = sB[buf];

      #pragma unroll
      for (int ks = 0; ks < KC_; ks += 32) {
        // A fragment (16x32 bf16, ISA 7.12.2): lane holds K = hl*8+[0..7]
        // (VGPR0-3) and K = 16+hl*8+[0..7] (VGPR4-7) of row m.
        Frag fa;
        const unsigned short* ap = cA + (wave * 16 + m) * ASTR + ks + hl * 8;
        fa.q[0] = *(const uint4*)ap;
        fa.q[1] = *(const uint4*)(ap + 16);
        // Load all B fragments ahead of their WMMA so ds latency overlaps.
        Frag fb[4];
        #pragma unroll
        for (int jj = 0; jj < 4; ++jj) {
          const unsigned short* bp = cB + (jj * 16 + m) * ASTR + ks + hl * 8;
          fb[jj].q[0] = *(const uint4*)bp;
          fb[jj].q[1] = *(const uint4*)(bp + 16);
        }
        #pragma unroll
        for (int jj = 0; jj < 4; ++jj)
          acc[jj] = __builtin_amdgcn_wmma_f32_16x16x32_bf16(
              false, fa.v, false, fb[jj].v, (short)0, acc[jj], false, false);
      }
    }

    // Epilogue: C layout -> VGPR r holds row (hl*8 + r), column = lane&15.
    #pragma unroll
    for (int jj = 0; jj < 4; ++jj) {
      const int j  = jo * 64 + jj * 16 + m;
      const int js = (j >= S_) ? j - S_ : j;
      const float mj = mb[js] ? 1.0f : 0.0f;
      #pragma unroll
      for (int r = 0; r < 8; ++r) {
        const int i  = rowStart + hl * 8 + r;
        const int is = (i >= S_) ? i - S_ : i;
        const float e = __expf(acc[jj][r] * INV_T);
        racc[r] += (is == js) ? 0.0f : mj * e;   // exact same-token exclusion
      }
    }
  }

  // Reduce each row sum across the 16 lanes of its half-wave.
  #pragma unroll
  for (int r = 0; r < 8; ++r) {
    float v = racc[r];
    v += __shfl_xor(v, 1);
    v += __shfl_xor(v, 2);
    v += __shfl_xor(v, 4);
    v += __shfl_xor(v, 8);
    racc[r] = v;
  }
  if (m == 0) {
    #pragma unroll
    for (int r = 0; r < 8; ++r)
      Ng[(size_t)b * N2_ + rowStart + hl * 8 + r] = racc[r];
  }
}

// ---------------------------------------------------------------------------
// Kernel 3: per-sample masked mean of -log(pos/(Ng+pos)).
// ---------------------------------------------------------------------------
__global__ __launch_bounds__(256) void per_sample_kernel(
    const float* __restrict__ Ng, const float* __restrict__ possim,
    const unsigned char* __restrict__ mask, float* __restrict__ per_sample) {
  const int b = blockIdx.x;
  const int t = threadIdx.x;
  float sum = 0.f, cnt = 0.f;
  for (int i = t; i < N2_; i += 256) {
    const int s = (i >= S_) ? i - S_ : i;
    if (mask[(size_t)b * S_ + s]) {
      const float ps = possim[(size_t)b * S_ + s];   // already /T
      const float p  = __expf(ps);
      sum += __logf(Ng[(size_t)b * N2_ + i] + p) - ps;
      cnt += 1.0f;
    }
  }
  #pragma unroll
  for (int off = 16; off; off >>= 1) {
    sum += __shfl_xor(sum, off);
    cnt += __shfl_xor(cnt, off);
  }
  __shared__ float rs[8], rc[8];
  const int w = t >> 5, l = t & 31;
  if (l == 0) { rs[w] = sum; rc[w] = cnt; }
  __syncthreads();
  if (t == 0) {
    float ssum = 0.f, scnt = 0.f;
    #pragma unroll
    for (int i = 0; i < 8; ++i) { ssum += rs[i]; scnt += rc[i]; }
    per_sample[b] = ssum / scnt;
  }
}

__global__ void final_kernel(const float* __restrict__ per_sample,
                             float* __restrict__ out) {
  if (threadIdx.x == 0 && blockIdx.x == 0) {
    float s = 0.f;
    #pragma unroll
    for (int i = 0; i < B_; ++i) s += per_sample[i];
    out[0] = s * (1.0f / B_);
  }
}

// ---------------------------------------------------------------------------
extern "C" void kernel_launch(void* const* d_in, const int* in_sizes, int n_in,
                              void* d_out, int out_size, void* d_ws, size_t ws_size,
                              hipStream_t stream) {
  const float* h1 = (const float*)d_in[0];
  const float* h2 = (const float*)d_in[1];
  const unsigned char* mask = (const unsigned char*)d_in[2];  // jnp bool = 1 byte
  float* out = (float*)d_out;

  // Workspace: bf16 features (48MB), Ng row sums, pos logits, per-sample.
  unsigned short* Xb = (unsigned short*)d_ws;
  const size_t xbytes = (size_t)B_ * N2_ * D_ * sizeof(unsigned short);
  float* Ng         = (float*)((char*)d_ws + xbytes);
  float* possim     = Ng + (size_t)B_ * N2_;
  float* per_sample = possim + (size_t)B_ * S_;

  norm_kernel<<<dim3(S_, B_), 256, 0, stream>>>(h1, h2, Xb, possim);
  gemm_rowsum_kernel<<<dim3(N2_ / 64, B_), 128, 0, stream>>>(Xb, mask, Ng);
  per_sample_kernel<<<dim3(B_), 256, 0, stream>>>(Ng, possim, mask, per_sample);
  final_kernel<<<1, 32, 0, stream>>>(per_sample, out);
}
